// KSpaceTransformerGNNEncoder_31714038514066
// MI455X (gfx1250) — matrix operand order repaired and checked by reading, not compiled, not run on hardware
//
#include <hip/hip_runtime.h>

// ---------------------------------------------------------------------------
// Types for CDNA5 WMMA (gfx1250, wave32)
// ---------------------------------------------------------------------------
typedef __attribute__((ext_vector_type(16))) __bf16 v16bf;
typedef __attribute__((ext_vector_type(8)))  float  v8f;
typedef unsigned short ushort_t;

union Frag16 { v16bf v; ushort_t s[16]; };

// ---------------------------------------------------------------------------
// Problem constants (match reference)
// ---------------------------------------------------------------------------
#define N_NODES 20000
#define N_EDGES 320000
#define N_GRAPH 64
#define F_IN    128
#define H_DIM   256
#define HEADS   8
#define HEAD_C  32
#define LAYERS  8
#define F_OUT   128
#define ATT_SCALE 0.17677669529663687f  // 1/sqrt(32)

__device__ __forceinline__ ushort_t f2bf(float f) {
  union { float f; unsigned u; } v; v.f = f;
  unsigned x = v.u;
  x += 0x7FFFu + ((x >> 16) & 1u);          // round-to-nearest-even
  return (ushort_t)(x >> 16);
}

__device__ __forceinline__ void atomicMaxFloat(float* addr, float val) {
  if (val >= 0.0f) atomicMax((int*)addr, __float_as_int(val));
  else             atomicMin((unsigned int*)addr, __float_as_uint(val));
}

__device__ __forceinline__ float block_reduce_sum256(float v, float* sdata) {
  int tid = threadIdx.x;
  sdata[tid] = v;
  __syncthreads();
  for (int s = 128; s > 0; s >>= 1) {
    if (tid < s) sdata[tid] += sdata[tid + s];
    __syncthreads();
  }
  float r = sdata[0];
  __syncthreads();
  return r;
}

// ---------------------------------------------------------------------------
// bf16 WMMA GEMM with pre-packed B (fragment-order weights):
//   C[M,Nout] = A[M,K] @ B[K,Nout] + bias
//   grid = (Nout/128, M/32), block = 256 threads (8 waves).
//   Each wave owns a 16-col tile and TWO 16-row tiles (B frag reused).
//   Packed B layout: Bp[((ct*NKS + ks)*32 + lane)*16 + h]
//     where col = ct*16 + (lane&15), k = ks*32 + (h<8?0:8) + (lane<16?0:8) + h.
//   K is a template constant -> K-loop fully unrolls into back-to-back WMMAs.
// ---------------------------------------------------------------------------
template <int K>
__global__ void __launch_bounds__(256)
k_gemm_bf16_wmma(const ushort_t* __restrict__ A,
                 const ushort_t* __restrict__ Bp,
                 const float* __restrict__ bias,
                 float* __restrict__ C,
                 int Nout)
{
  constexpr int NKS = K / 32;
  __shared__ ushort_t lds_a[32 * K];
  const int tid  = threadIdx.x;
  const int wave = tid >> 5;
  const int lane = tid & 31;
  const int row0 = blockIdx.y * 32;
  const int ctg  = blockIdx.x * 8 + wave;          // global 16-col tile index
  const int col  = ctg * 16 + (lane & 15);

  // Stage contiguous 32xK A tile into LDS with 128-bit copies
  const long long abase = (long long)row0 * K;
  for (int i = tid * 8; i < 32 * K; i += 256 * 8)
    *(float4*)&lds_a[i] = *(const float4*)&A[abase + i];
  __syncthreads();

  v8f acc0 = {}, acc1 = {};
  const int arow   = lane & 15;
  const int ka_off = (lane < 16) ? 0 : 8;
  const ushort_t* brow = Bp + ((long long)ctg * NKS * 32 + lane) * 16;

#pragma unroll
  for (int ks = 0; ks < NKS; ++ks) {
    Frag16 au0, au1, bu;
    const ushort_t* a0 = &lds_a[arow * K + ks * 32 + ka_off];
    const ushort_t* a1 = &lds_a[(arow + 16) * K + ks * 32 + ka_off];
    *(float4*)&au0.s[0] = *(const float4*)(a0);        // K = ks*32+ka_off+0..7
    *(float4*)&au0.s[8] = *(const float4*)(a0 + 16);   // K = ks*32+ka_off+16..23
    *(float4*)&au1.s[0] = *(const float4*)(a1);
    *(float4*)&au1.s[8] = *(const float4*)(a1 + 16);
    const ushort_t* bp = brow + (long long)ks * 32 * 16;  // 32B contiguous per lane
    *(float4*)&bu.s[0] = *(const float4*)(bp);
    *(float4*)&bu.s[8] = *(const float4*)(bp + 8);

    acc0 = __builtin_amdgcn_wmma_f32_16x16x32_bf16(
        false, au0.v, false, bu.v, (short)0, acc0, false, false);
    acc1 = __builtin_amdgcn_wmma_f32_16x16x32_bf16(
        false, au1.v, false, bu.v, (short)0, acc1, false, false);
  }

  const float bv = bias ? bias[col] : 0.0f;
  const int m0 = row0 + ((lane < 16) ? 0 : 8);
#pragma unroll
  for (int r = 0; r < 8; ++r) {
    C[(long long)(m0 + r) * Nout + col]      = acc0[r] + bv;
    C[(long long)(m0 + 16 + r) * Nout + col] = acc1[r] + bv;
  }
}

// ---------------------------------------------------------------------------
// Prep kernels
// ---------------------------------------------------------------------------
__global__ void k_f2bf(const float* __restrict__ src,
                       ushort_t* __restrict__ dst, long long n) {
  long long t = (long long)blockIdx.x * blockDim.x + threadIdx.x;
  if (t < n) dst[t] = f2bf(src[t]);
}

// Pack a [K,Nout] f32 weight into bf16 WMMA-fragment order (see layout above).
__global__ void k_pack_b(const float* __restrict__ W, ushort_t* __restrict__ Bp,
                         int K, int Nout) {
  int total = K * Nout;
  int t = blockIdx.x * blockDim.x + threadIdx.x;
  if (t >= total) return;
  int nks  = K / 32;
  int h    = t & 15;
  int lane = (t >> 4) & 31;
  int ks   = (t >> 9) % nks;
  int ct   = t / (512 * nks);
  int col  = ct * 16 + (lane & 15);
  int k    = ks * 32 + ((h < 8) ? 0 : 8) + ((lane < 16) ? 0 : 8) + h;
  Bp[t] = f2bf(W[(long long)k * Nout + col]);
}

// Fused per-layer weight concat [256,1024] (q|k|v|s) packed to fragment order.
__global__ void k_pack_wcat(const float* __restrict__ Wq, const float* __restrict__ Wk,
                            const float* __restrict__ Wv, const float* __restrict__ Ws,
                            ushort_t* __restrict__ wcat) {
  long long t = (long long)blockIdx.x * blockDim.x + threadIdx.x;
  if (t >= (long long)LAYERS * H_DIM * 4 * H_DIM) return;
  int l = (int)(t >> 18);              // 256*1024 = 2^18 per layer
  int p = (int)(t & 262143);
  int h    = p & 15;
  int lane = (p >> 4) & 31;
  int ks   = (p >> 9) & 7;             // NKS = 8 for K=256
  int ct   = p >> 12;                  // 64 col tiles for Nout=1024
  int col  = ct * 16 + (lane & 15);
  int k    = ks * 32 + ((h < 8) ? 0 : 8) + ((lane < 16) ? 0 : 8) + h;
  int sel = col >> 8, jj = col & 255;
  const float* W = (sel == 0) ? Wq : (sel == 1) ? Wk : (sel == 2) ? Wv : Ws;
  wcat[t] = f2bf(W[(long long)l * H_DIM * H_DIM + k * H_DIM + jj]);
}

__global__ void k_build_bcat(const float* __restrict__ bq, const float* __restrict__ bk,
                             const float* __restrict__ bv, const float* __restrict__ bs,
                             float* __restrict__ bcat) {
  int t = blockIdx.x * blockDim.x + threadIdx.x;
  if (t >= LAYERS * 4 * H_DIM) return;
  int l = t >> 10, j = t & 1023;
  int sel = j >> 8, jj = j & 255;
  const float* b = (sel == 0) ? bq : (sel == 1) ? bk : (sel == 2) ? bv : bs;
  bcat[t] = b[l * H_DIM + jj];
}

// ---------------------------------------------------------------------------
// Attention kernels
// ---------------------------------------------------------------------------
__global__ void k_attn_init(float* __restrict__ amax, float* __restrict__ denom,
                            float* __restrict__ agg) {
  long long t = (long long)blockIdx.x * blockDim.x + threadIdx.x;
  if (t < (long long)N_NODES * H_DIM) agg[t] = 0.0f;
  if (t < (long long)N_NODES * HEADS) { amax[t] = -1e30f; denom[t] = 0.0f; }
}

__global__ void k_edge_alpha(const float* __restrict__ qkvs, const int* __restrict__ ei,
                             float* __restrict__ alpha, float* __restrict__ amax) {
  long long t = (long long)blockIdx.x * blockDim.x + threadIdx.x;
  if (t >= (long long)N_EDGES * HEADS) return;
  int e = (int)(t >> 3), hd = (int)(t & 7);
  int src = ei[e], dst = ei[N_EDGES + e];
  const float4* q = (const float4*)(qkvs + (long long)dst * 1024 + hd * HEAD_C);
  const float4* k = (const float4*)(qkvs + (long long)src * 1024 + H_DIM + hd * HEAD_C);
  float s = 0.0f;
#pragma unroll
  for (int c = 0; c < HEAD_C / 4; ++c) {
    float4 qv = q[c], kv = k[c];
    s += qv.x * kv.x + qv.y * kv.y + qv.z * kv.z + qv.w * kv.w;
  }
  s *= ATT_SCALE;
  alpha[t] = s;
  atomicMaxFloat(&amax[dst * HEADS + hd], s);
}

__global__ void k_edge_exp(const int* __restrict__ ei, float* __restrict__ alpha,
                           const float* __restrict__ amax, float* __restrict__ denom) {
  long long t = (long long)blockIdx.x * blockDim.x + threadIdx.x;
  if (t >= (long long)N_EDGES * HEADS) return;
  int e = (int)(t >> 3), hd = (int)(t & 7);
  int dst = ei[N_EDGES + e];
  float ex = __expf(alpha[t] - amax[dst * HEADS + hd]);
  alpha[t] = ex;
  atomicAdd(&denom[dst * HEADS + hd], ex);
}

// One thread per (edge, 4 channels): float4 value load + 4 f32 atomics.
__global__ void k_edge_agg(const float* __restrict__ qkvs, const int* __restrict__ ei,
                           const float* __restrict__ exv, const float* __restrict__ denom,
                           float* __restrict__ agg) {
  long long t = (long long)blockIdx.x * blockDim.x + threadIdx.x;
  if (t >= (long long)N_EDGES * (H_DIM / 4)) return;
  int e = (int)(t >> 6), c = (int)(t & 63) * 4, hd = c >> 5;
  int src = ei[e], dst = ei[N_EDGES + e];
  float attn = exv[(long long)e * HEADS + hd] / (denom[dst * HEADS + hd] + 1e-16f);
  float4 vv = *(const float4*)(qkvs + (long long)src * 1024 + 2 * H_DIM + c);
  float* a = agg + (long long)dst * H_DIM + c;
  atomicAdd(a + 0, vv.x * attn);
  atomicAdd(a + 1, vv.y * attn);
  atomicAdd(a + 2, vv.z * attn);
  atomicAdd(a + 3, vv.w * attn);
}

// ---------------------------------------------------------------------------
// Node update: beta-gated skip + LayerNorm + ReLU. One block (256 thr) / node.
// ---------------------------------------------------------------------------
__global__ void __launch_bounds__(256)
k_node_update(const float* __restrict__ agg, const float* __restrict__ qkvs,
              const float* __restrict__ wbeta, const float* __restrict__ lng,
              const float* __restrict__ lnb,
              float* __restrict__ h32, ushort_t* __restrict__ h16) {
  __shared__ float sdata[256];
  int n = blockIdx.x, c = threadIdx.x;
  float o  = agg[(long long)n * H_DIM + c];
  float xr = qkvs[(long long)n * 1024 + 3 * H_DIM + c];
  float pb = o * wbeta[c] + xr * wbeta[H_DIM + c] + (o - xr) * wbeta[2 * H_DIM + c];
  float bsum = block_reduce_sum256(pb, sdata);
  float beta = 1.0f / (1.0f + __expf(-bsum));
  float hc = beta * xr + (1.0f - beta) * o;
  float mu = block_reduce_sum256(hc, sdata) * (1.0f / H_DIM);
  float d = hc - mu;
  float var = block_reduce_sum256(d * d, sdata) * (1.0f / H_DIM);
  float y = d * rsqrtf(var + 1e-5f) * lng[c] + lnb[c];
  y = fmaxf(y, 0.0f);
  h32[(long long)n * H_DIM + c] = y;
  h16[(long long)n * H_DIM + c] = f2bf(y);
}

// ---------------------------------------------------------------------------
// Pooling
// ---------------------------------------------------------------------------
__global__ void k_pool_init(float* __restrict__ psum, float* __restrict__ pcnt) {
  int t = blockIdx.x * blockDim.x + threadIdx.x;
  if (t < N_GRAPH * H_DIM) psum[t] = 0.0f;
  if (t < N_GRAPH) pcnt[t] = 0.0f;
}

__global__ void k_pool_add(const float* __restrict__ h32, const int* __restrict__ batch,
                           float* __restrict__ psum, float* __restrict__ pcnt) {
  long long t = (long long)blockIdx.x * blockDim.x + threadIdx.x;
  if (t >= (long long)N_NODES * H_DIM) return;
  int n = (int)(t >> 8), c = (int)(t & 255);
  int g = batch[n];
  atomicAdd(&psum[g * H_DIM + c], h32[t]);
  if (c == 0) atomicAdd(&pcnt[g], 1.0f);
}

__global__ void k_pool_fin(const float* __restrict__ psum, const float* __restrict__ pcnt,
                           ushort_t* __restrict__ p16) {
  int t = blockIdx.x * blockDim.x + threadIdx.x;
  if (t >= N_GRAPH * H_DIM) return;
  int g = t >> 8;
  p16[t] = f2bf(psum[t] / fmaxf(pcnt[g], 1.0f));
}

// ---------------------------------------------------------------------------
// Host orchestration
// ---------------------------------------------------------------------------
extern "C" void kernel_launch(void* const* d_in, const int* in_sizes, int n_in,
                              void* d_out, int out_size, void* d_ws, size_t ws_size,
                              hipStream_t stream) {
  (void)in_sizes; (void)n_in; (void)out_size; (void)ws_size;
  const float* x       = (const float*)d_in[0];
  const int*   ei      = (const int*)d_in[1];
  const int*   batch   = (const int*)d_in[2];
  const float* W_init  = (const float*)d_in[3];
  const float* b_init  = (const float*)d_in[4];
  const float* Wq      = (const float*)d_in[5];
  const float* bq      = (const float*)d_in[6];
  const float* Wk      = (const float*)d_in[7];
  const float* bk      = (const float*)d_in[8];
  const float* Wv      = (const float*)d_in[9];
  const float* bv      = (const float*)d_in[10];
  const float* Ws      = (const float*)d_in[11];
  const float* bs      = (const float*)d_in[12];
  const float* Wbeta   = (const float*)d_in[13];
  const float* ln_g    = (const float*)d_in[14];
  const float* ln_b    = (const float*)d_in[15];
  const float* W_final = (const float*)d_in[16];
  const float* b_final = (const float*)d_in[17];
  float* out = (float*)d_out;

  // ---- workspace bump allocator (256B aligned) ----
  char* base = (char*)d_ws;
  size_t off = 0;
  auto ALLOC = [&](size_t bytes) -> void* {
    off = (off + 255) & ~(size_t)255;
    void* p = base + off;
    off += bytes;
    return p;
  };
  float*    h32   = (float*)ALLOC(sizeof(float) * (size_t)N_NODES * H_DIM);
  ushort_t* h16   = (ushort_t*)ALLOC(2ull * N_NODES * H_DIM);
  float*    qkvs  = (float*)ALLOC(sizeof(float) * (size_t)N_NODES * 4 * H_DIM);
  float*    exbuf = (float*)ALLOC(sizeof(float) * (size_t)N_EDGES * HEADS);
  float*    amax  = (float*)ALLOC(sizeof(float) * (size_t)N_NODES * HEADS);
  float*    denom = (float*)ALLOC(sizeof(float) * (size_t)N_NODES * HEADS);
  float*    agg   = (float*)ALLOC(sizeof(float) * (size_t)N_NODES * H_DIM);
  ushort_t* wcat  = (ushort_t*)ALLOC(2ull * LAYERS * H_DIM * 4 * H_DIM);
  float*    bcat  = (float*)ALLOC(sizeof(float) * LAYERS * 4 * H_DIM);
  ushort_t* wi16  = (ushort_t*)ALLOC(2ull * F_IN * H_DIM);
  ushort_t* wf16  = (ushort_t*)ALLOC(2ull * H_DIM * F_OUT);
  ushort_t* x16   = (ushort_t*)ALLOC(2ull * N_NODES * F_IN);
  float*    psum  = (float*)ALLOC(sizeof(float) * N_GRAPH * H_DIM);
  float*    pcnt  = (float*)ALLOC(sizeof(float) * N_GRAPH);
  ushort_t* p16   = (ushort_t*)ALLOC(2ull * N_GRAPH * H_DIM);

  auto cdiv = [](long long a, long long b) { return (unsigned)((a + b - 1) / b); };

  // ---- prep: bf16 conversions + fragment-order weight packing ----
  k_f2bf<<<cdiv((long long)N_NODES * F_IN, 256), 256, 0, stream>>>(x, x16, (long long)N_NODES * F_IN);
  k_pack_b<<<cdiv(F_IN * H_DIM, 256), 256, 0, stream>>>(W_init, wi16, F_IN, H_DIM);
  k_pack_b<<<cdiv(H_DIM * F_OUT, 256), 256, 0, stream>>>(W_final, wf16, H_DIM, F_OUT);
  k_pack_wcat<<<cdiv((long long)LAYERS * H_DIM * 4 * H_DIM, 256), 256, 0, stream>>>(Wq, Wk, Wv, Ws, wcat);
  k_build_bcat<<<cdiv(LAYERS * 4 * H_DIM, 256), 256, 0, stream>>>(bq, bk, bv, bs, bcat);

  // ---- initial projection: h = x @ W_init + b_init  (K=128) ----
  k_gemm_bf16_wmma<F_IN><<<dim3(H_DIM / 128, N_NODES / 32), 256, 0, stream>>>(
      x16, wi16, b_init, h32, H_DIM);
  k_f2bf<<<cdiv((long long)N_NODES * H_DIM, 256), 256, 0, stream>>>(h32, h16, (long long)N_NODES * H_DIM);

  // ---- 8 TransformerConv layers ----
  for (int l = 0; l < LAYERS; ++l) {
    k_gemm_bf16_wmma<H_DIM><<<dim3(4 * H_DIM / 128, N_NODES / 32), 256, 0, stream>>>(
        h16, wcat + (size_t)l * H_DIM * 4 * H_DIM, bcat + l * 4 * H_DIM, qkvs, 4 * H_DIM);
    k_attn_init<<<cdiv((long long)N_NODES * H_DIM, 256), 256, 0, stream>>>(amax, denom, agg);
    k_edge_alpha<<<cdiv((long long)N_EDGES * HEADS, 256), 256, 0, stream>>>(qkvs, ei, exbuf, amax);
    k_edge_exp<<<cdiv((long long)N_EDGES * HEADS, 256), 256, 0, stream>>>(ei, exbuf, amax, denom);
    k_edge_agg<<<cdiv((long long)N_EDGES * (H_DIM / 4), 256), 256, 0, stream>>>(qkvs, ei, exbuf, denom, agg);
    k_node_update<<<N_NODES, 256, 0, stream>>>(
        agg, qkvs, Wbeta + (size_t)l * 3 * H_DIM, ln_g + l * H_DIM, ln_b + l * H_DIM, h32, h16);
  }

  // ---- global mean pool + final projection (K=256) ----
  k_pool_init<<<cdiv(N_GRAPH * H_DIM, 256), 256, 0, stream>>>(psum, pcnt);
  k_pool_add<<<cdiv((long long)N_NODES * H_DIM, 256), 256, 0, stream>>>(h32, batch, psum, pcnt);
  k_pool_fin<<<cdiv(N_GRAPH * H_DIM, 256), 256, 0, stream>>>(psum, pcnt, p16);
  k_gemm_bf16_wmma<H_DIM><<<dim3(F_OUT / 128, N_GRAPH / 32), 256, 0, stream>>>(
      p16, wf16, b_final, out, F_OUT);
}
